// SelfAttention_20160576488071
// MI455X (gfx1250) — matrix operand order repaired and checked by reading, not compiled, not run on hardware
//
#include <hip/hip_runtime.h>

// ---------------- problem constants ----------------
#define EMBED 1024
#define HEADS 16
#define HD    64
#define NB    8
#define SEQL  1024

// ---------------- types ----------------
typedef __bf16 bf16;
typedef bf16  v2bf  __attribute__((ext_vector_type(2)));
typedef bf16  v8bf  __attribute__((ext_vector_type(8)));
typedef bf16  v16bf __attribute__((ext_vector_type(16)));
typedef float v8f   __attribute__((ext_vector_type(8)));

// packed f32x2 -> bf16x2: single v_cvt_pk_bf16_f32 (gfx1250 bf16 VALU set).
static __device__ __forceinline__ v2bf f2bf2(float a, float b) {
#if __has_builtin(__builtin_amdgcn_cvt_pk_bf16_f32)
    return __builtin_amdgcn_cvt_pk_bf16_f32(a, b);
#else
    unsigned r;
    asm("v_cvt_pk_bf16_f32 %0, %1, %2" : "=v"(r) : "v"(a), "v"(b));
    v2bf out; __builtin_memcpy(&out, &r, 4);
    return out;
#endif
}

static __device__ __forceinline__ v16bf cat8(v8bf lo, v8bf hi) {
    return __builtin_shufflevector(lo, hi, 0,1,2,3,4,5,6,7,8,9,10,11,12,13,14,15);
}

// A-fragment (16xK row-major source, this lane = row lane%16).
// ISA 7.12.2 16-bit A 16x32: lane-half h holds K = {kb+8h..kb+8h+7, kb+16+8h..kb+16+8h+7}
static __device__ __forceinline__ v16bf ldA(const bf16* rowptr, int kb, int half) {
    v8bf lo = *(const v8bf*)(rowptr + kb + 8 * half);
    v8bf hi = *(const v8bf*)(rowptr + kb + 16 + 8 * half);
    return cat8(lo, hi);
}

// B-fragment (KxN, source stored [col][k] contiguous; this lane = col lane%16).
// lane-half h holds K = {kb+8h..kb+8h+7, kb+16+8h..kb+16+8h+7} (same as A)
static __device__ __forceinline__ v16bf ldB(const bf16* colptr, int kb, int half) {
    v8bf lo = *(const v8bf*)(colptr + kb + 8 * half);
    v8bf hi = *(const v8bf*)(colptr + kb + 16 + 8 * half);
    return cat8(lo, hi);
}

static __device__ __forceinline__ v8f wmma_bf16(v16bf a, v16bf b, v8f c) {
    // (neg_a, A, neg_b, B, c_mod, C, reuse_a, reuse_b)
    return __builtin_amdgcn_wmma_f32_16x16x32_bf16(false, a, false, b, (short)0, c, false, false);
}

// =====================================================================
// Kernel 1: per-head projection  out[nh][row][col] = sum_d X[row,d]*W[col,d]
// grid = (SEQ/128, N*HEADS), block = 256 (8 waves, 16 rows each)
// =====================================================================
__global__ __launch_bounds__(256) void proj_bf16_kernel(
    const float* __restrict__ src, const float* __restrict__ W, bf16* __restrict__ dst)
{
    __shared__ alignas(16) bf16 Wl[64 * 64];     // [col][k]  == B layout
    __shared__ alignas(16) bf16 Xl[128 * 64];    // [row][k]  == A layout

    const int nh   = blockIdx.y;
    const int n    = nh / HEADS, h = nh % HEADS;
    const int rblk = blockIdx.x;
    const int t    = threadIdx.x;

    // W: packed convert, 32-bit LDS stores
    for (int i = 2 * t; i < 64 * 64; i += 512)
        *(v2bf*)&Wl[i] = f2bf2(W[i], W[i + 1]);

    const float* xs = src + (size_t)n * SEQL * EMBED + (size_t)(rblk * 128) * EMBED + h * HD;
    for (int i = 2 * t; i < 128 * 64; i += 512) {
        int r = i >> 6, d = i & 63;               // 2 | 64 -> pair never crosses a row
        *(v2bf*)&Xl[i] = f2bf2(xs[(size_t)r * EMBED + d], xs[(size_t)r * EMBED + d + 1]);
    }
    __syncthreads();

    const int wave = t >> 5, lane = t & 31;
    const int lrow = lane & 15, half = lane >> 4;
    const int rbase = wave * 16;

    v8f acc[4] = {};
    #pragma unroll
    for (int k0 = 0; k0 < 64; k0 += 32) {
        v16bf a = ldA(&Xl[(rbase + lrow) * 64], k0, half);
        #pragma unroll
        for (int c = 0; c < 4; ++c) {
            v16bf b = ldB(&Wl[(c * 16 + lrow) * 64], k0, half);
            acc[c] = wmma_bf16(a, b, acc[c]);
        }
    }

    bf16* dp = dst + (size_t)nh * SEQL * HD + (size_t)(rblk * 128) * HD;
    #pragma unroll
    for (int i = 0; i < 8; ++i) {
        int row = rbase + i + 8 * half;           // C/D layout: VGPR i -> row i + 8*half
        v2bf pkA = f2bf2(acc[0][i], acc[1][i]);
        v2bf pkB = f2bf2(acc[2][i], acc[3][i]);
        dp[row * HD +  0 + lrow] = pkA[0];
        dp[row * HD + 16 + lrow] = pkA[1];
        dp[row * HD + 32 + lrow] = pkB[0];
        dp[row * HD + 48 + lrow] = pkB[1];
    }
}

// =====================================================================
// Kernel 2: flash attention per (n,h).  grid = (SEQ/128, N*HEADS), block = 256
// Double-buffered V^T staging: one barrier per 32-key block.
// =====================================================================
__global__ __launch_bounds__(256) void attn_flash_kernel(
    const bf16* __restrict__ Qp, const bf16* __restrict__ Kp, const bf16* __restrict__ Vp,
    const int* __restrict__ mask, bf16* __restrict__ ctx)
{
    __shared__ alignas(16) bf16 Vt[2][64 * 32];    // [buf][d][key]  transposed V block
    __shared__ alignas(16) bf16 Pl[8][16 * 32];    // per-wave P tile [row][key]

    const int nh   = blockIdx.y;
    const int n    = nh / HEADS, h = nh % HEADS;
    const int qblk = blockIdx.x;
    const int t    = threadIdx.x, wave = t >> 5, lane = t & 31;
    const int lrow = lane & 15, half = lane >> 4;

    const bf16* q  = Qp + (size_t)nh * SEQL * HD + (size_t)(qblk * 128 + wave * 16) * HD;
    const bf16* kb = Kp + (size_t)nh * SEQL * HD;
    const bf16* vb = Vp + (size_t)nh * SEQL * HD;
    const int*  mr = mask + (size_t)n * SEQL;

    // Q tile A-fragments, K = 64 -> two k-steps, held in registers for the whole loop
    v16bf qa0 = ldA(q + lrow * HD, 0,  half);
    v16bf qa1 = ldA(q + lrow * HD, 32, half);

    float mreg[8], lreg[8];
    v8f o[4] = {};
    #pragma unroll
    for (int i = 0; i < 8; ++i) { mreg[i] = -3.0e38f; lreg[i] = 0.0f; }
    const float scale = 0.03125f;   // 1/sqrt(EMBED)

    int buf = 0;
    for (int jb = 0; jb < SEQL; jb += 32, buf ^= 1) {
        { // stage V^T : Vt[buf][d][key] = Vp[jb+key][d]   (buf last read 2 iters ago)
            int key = t >> 3;
            int d0  = (t & 7) * 8;
            v8bf vv = *(const v8bf*)(vb + (size_t)(jb + key) * HD + d0);
            #pragma unroll
            for (int j = 0; j < 8; ++j) Vt[buf][(d0 + j) * 32 + key] = vv[j];
        }
        // prefetch next K/V blocks into cache while this block computes
        if (jb + 32 < SEQL) {
            __builtin_prefetch(kb + (size_t)(jb + 32 + lane) * HD, 0, 1);
            __builtin_prefetch(vb + (size_t)(jb + 32 + lane) * HD, 0, 1);
        }
        __syncthreads();

        // S = Q x K^T for 32 keys (two 16-col C tiles); K rows are already B-layout
        v8f s0 = {}, s1 = {};
        {
            const bf16* kr0 = kb + (size_t)(jb + lrow) * HD;
            const bf16* kr1 = kb + (size_t)(jb + 16 + lrow) * HD;
            s0 = wmma_bf16(qa0, ldB(kr0, 0, half), s0);
            s0 = wmma_bf16(qa1, ldB(kr0, 32, half), s0);
            s1 = wmma_bf16(qa0, ldB(kr1, 0, half), s1);
            s1 = wmma_bf16(qa1, ldB(kr1, 32, half), s1);
        }

        const int m0 = mr[jb + lrow], m1 = mr[jb + 16 + lrow];
        #pragma unroll
        for (int i = 0; i < 8; ++i) {
            float a0 = m0 ? s0[i] * scale : -1.0e10f;
            float a1 = m1 ? s1[i] * scale : -1.0e10f;
            // row max across the 16 lanes holding this row
            float v = fmaxf(a0, a1);
            for (int d = 1; d < 16; d <<= 1) v = fmaxf(v, __shfl_xor(v, d, 32));
            float mn   = fmaxf(mreg[i], v);
            float corr = __expf(mreg[i] - mn);
            float p0   = __expf(a0 - mn);
            float p1   = __expf(a1 - mn);
            float ps   = p0 + p1;
            for (int d = 1; d < 16; d <<= 1) ps += __shfl_xor(ps, d, 32);
            lreg[i] = lreg[i] * corr + ps;
            mreg[i] = mn;
            #pragma unroll
            for (int c = 0; c < 4; ++c) o[c][i] *= corr;
            // C-layout -> A-layout via per-wave LDS tile (same-wave LDS is in-order)
            int row = i + 8 * half;
            v2bf pk = f2bf2(p0, p1);
            Pl[wave][row * 32 + lrow]      = pk[0];
            Pl[wave][row * 32 + 16 + lrow] = pk[1];
        }

        // O += P(16x32) x V(32x64)
        v16bf pa = ldA(&Pl[wave][lrow * 32], 0, half);
        #pragma unroll
        for (int c = 0; c < 4; ++c) {
            v16bf b = ldB(&Vt[buf][(c * 16 + lrow) * 32], 0, half);
            o[c] = wmma_bf16(pa, b, o[c]);
        }
    }

    // normalize and write ctx[n][q][h*64 + d] (bf16)
    bf16* cp = ctx + (size_t)n * SEQL * EMBED + (size_t)(qblk * 128 + wave * 16) * EMBED + h * HD;
    #pragma unroll
    for (int i = 0; i < 8; ++i) {
        float inv = 1.0f / lreg[i];
        int row = i + 8 * half;
        v2bf pkA = f2bf2(o[0][i] * inv, o[1][i] * inv);
        v2bf pkB = f2bf2(o[2][i] * inv, o[3][i] * inv);
        cp[(size_t)row * EMBED +  0 + lrow] = pkA[0];
        cp[(size_t)row * EMBED + 16 + lrow] = pkA[1];
        cp[(size_t)row * EMBED + 32 + lrow] = pkB[0];
        cp[(size_t)row * EMBED + 48 + lrow] = pkB[1];
    }
}

// =====================================================================
// Kernel 3: out = ctx @ Wo^T + bo.  grid = (8192/128, 1024/64), block = 256
// =====================================================================
__global__ __launch_bounds__(256) void out_proj_kernel(
    const bf16* __restrict__ ctx, const float* __restrict__ Wo,
    const float* __restrict__ bo, float* __restrict__ out)
{
    __shared__ alignas(16) bf16 Xl[128 * 32];   // [row][k]
    __shared__ alignas(16) bf16 Wt[64 * 32];    // [col][k]  (Wo rows are already [e][k])

    const int rblk = blockIdx.x, cblk = blockIdx.y;
    const int t = threadIdx.x, wave = t >> 5, lane = t & 31;
    const int lrow = lane & 15, half = lane >> 4;

    const bf16*  xs   = ctx + (size_t)(rblk * 128) * EMBED;
    const float* wsrc = Wo + (size_t)(cblk * 64) * EMBED;

    v8f acc[4] = {};
    for (int k0 = 0; k0 < EMBED; k0 += 32) {
        __syncthreads();
        { // stage A tile: 128 x 32 bf16 (copy)
            int r = t >> 1, p = (t & 1) * 16;
            const v8bf* sp = (const v8bf*)(xs + (size_t)r * EMBED + k0 + p);
            *(v8bf*)&Xl[r * 32 + p]     = sp[0];
            *(v8bf*)&Xl[r * 32 + p + 8] = sp[1];
        }
        { // stage B tile: 64 x 32, packed f32 -> bf16
            int e = t >> 2, p = (t & 3) * 8;
            const float* wp = wsrc + (size_t)e * EMBED + k0 + p;
            #pragma unroll
            for (int j = 0; j < 4; ++j)
                *(v2bf*)&Wt[e * 32 + p + 2 * j] = f2bf2(wp[2 * j], wp[2 * j + 1]);
        }
        __syncthreads();

        v16bf a = ldA(&Xl[(wave * 16 + lrow) * 32], 0, half);
        #pragma unroll
        for (int c = 0; c < 4; ++c) {
            v16bf b = ldB(&Wt[(c * 16 + lrow) * 32], 0, half);
            acc[c] = wmma_bf16(a, b, acc[c]);
        }
    }

    float* op = out + (size_t)(rblk * 128 + wave * 16) * EMBED + cblk * 64;
    #pragma unroll
    for (int c = 0; c < 4; ++c) {
        int col = c * 16 + lrow;
        float bias = bo[cblk * 64 + col];
        #pragma unroll
        for (int i = 0; i < 8; ++i)
            op[(size_t)(i + 8 * half) * EMBED + col] = acc[c][i] + bias;
    }
}

// =====================================================================
extern "C" void kernel_launch(void* const* d_in, const int* in_sizes, int n_in,
                              void* d_out, int out_size, void* d_ws, size_t ws_size,
                              hipStream_t stream) {
    const float* values = (const float*)d_in[0];
    const float* keys   = (const float*)d_in[1];
    const float* query  = (const float*)d_in[2];
    const int*   mask   = (const int*)d_in[3];
    const float* Wv     = (const float*)d_in[4];
    const float* Wk     = (const float*)d_in[5];
    const float* Wq     = (const float*)d_in[6];
    const float* Wo     = (const float*)d_in[7];
    const float* bo     = (const float*)d_in[8];
    float*       out    = (float*)d_out;

    // workspace layout (bf16): Qp | Kp | Vp | ctx  — 4 x 16 MB = 64 MB
    const size_t PROJ_BYTES = (size_t)NB * HEADS * SEQL * HD * sizeof(bf16);
    char* ws  = (char*)d_ws;
    bf16* Qp  = (bf16*)(ws);
    bf16* Kp  = (bf16*)(ws + PROJ_BYTES);
    bf16* Vp  = (bf16*)(ws + 2 * PROJ_BYTES);
    bf16* ctx = (bf16*)(ws + 3 * PROJ_BYTES);

    dim3 blk(256);
    dim3 g1(SEQL / 128, NB * HEADS);
    proj_bf16_kernel<<<g1, blk, 0, stream>>>(query,  Wq, Qp);
    proj_bf16_kernel<<<g1, blk, 0, stream>>>(keys,   Wk, Kp);
    proj_bf16_kernel<<<g1, blk, 0, stream>>>(values, Wv, Vp);

    attn_flash_kernel<<<g1, blk, 0, stream>>>(Qp, Kp, Vp, mask, ctx);

    dim3 g3(NB * SEQL / 128, EMBED / 64);
    out_proj_kernel<<<g3, blk, 0, stream>>>(ctx, Wo, bo, out);
}
